// MixtralSparseMoeBlock_36996848287848
// MI455X (gfx1250) — matrix operand order repaired and checked by reading, not compiled, not run on hardware
//
#include <hip/hip_runtime.h>
#include <hip/hip_bf16.h>

// Problem constants (from reference setup_inputs)
#define B_  2
#define S_  2048
#define H_  1024
#define F_  4096
#define E_  8
#define T_  (B_ * S_)          // 4096 tokens
#define MAXP T_                // padded per-expert segment capacity
#define MT  32                 // tokens per FFN workgroup tile
#define FC  128                // F-dimension chunk per iteration (1 N-subtile/wave)
#define HP  (H_ + 8)           // padded LDS row stride (bank-conflict avoidance)
#define FCP (FC + 8)

typedef __attribute__((ext_vector_type(16))) __bf16 v16bf;
typedef __attribute__((ext_vector_type(8)))  __bf16 v8bf;
typedef __attribute__((ext_vector_type(8)))  float  v8f;

// ---------------------------------------------------------------------------
// A-fragment (16x32 bf16) from LDS, row-major with stride ldk (elements).
// Layout: lanes 0-15 -> K0..15 (row M=lane);
//         lanes 16-31 -> K16..31 (matches B fragment).
// ---------------------------------------------------------------------------
__device__ __forceinline__ v16bf load_a_frag(const __bf16* base, int ldk, int k, int lane) {
  const int m = lane & 15;
  const int half = lane >> 4;
  const __bf16* p = base + m * ldk + k + half * 16;
  v8bf lo = *(const v8bf*)p;          // K = k + 16*half + 0..7
  v8bf hi = *(const v8bf*)(p + 8);    // K = k + 16*half + 8..15
  return __builtin_shufflevector(lo, hi, 0, 1, 2, 3, 4, 5, 6, 7,
                                 8, 9, 10, 11, 12, 13, 14, 15);
}

// ---------------------------------------------------------------------------
// B-fragment (32x16 bf16): lane n = lane&15 selects the N row of W (row-major
// [N,K] weight); 16 consecutive fp32 K values converted to bf16.
// lanes 0-15 hold K=0..15, lanes 16-31 hold K=16..31 (caller adds half*16).
// ---------------------------------------------------------------------------
__device__ __forceinline__ v16bf cvt_b_frag(const float* p) {
  const float4* q = (const float4*)p;
  float4 a = q[0], b = q[1], c = q[2], d = q[3];
  v16bf r;
  r[0]  = (__bf16)a.x; r[1]  = (__bf16)a.y; r[2]  = (__bf16)a.z; r[3]  = (__bf16)a.w;
  r[4]  = (__bf16)b.x; r[5]  = (__bf16)b.y; r[6]  = (__bf16)b.z; r[7]  = (__bf16)b.w;
  r[8]  = (__bf16)c.x; r[9]  = (__bf16)c.y; r[10] = (__bf16)c.z; r[11] = (__bf16)c.w;
  r[12] = (__bf16)d.x; r[13] = (__bf16)d.y; r[14] = (__bf16)d.z; r[15] = (__bf16)d.w;
  return r;
}

__device__ __forceinline__ v8f wmma_bf16(v16bf a, v16bf b, v8f c) {
  return __builtin_amdgcn_wmma_f32_16x16x32_bf16(
      /*neg_a=*/false, a, /*neg_b=*/false, b,
      /*c_mod=*/(short)0, c, /*reuse_a=*/false, /*reuse_b=*/false);
}

// ---------------------------------------------------------------------------
// Kernel 0: zero output (y + loss slot) and the ws header (cursors + sumprob).
// ---------------------------------------------------------------------------
__global__ void moe_zero_kernel(float* __restrict__ out, int n, float* __restrict__ wshdr) {
  int i = blockIdx.x * 256 + threadIdx.x;
  if (i < n) out[i] = 0.0f;
  if (i < 16) wshdr[i] = 0.0f;   // zero bits valid for both int cursors and float sums
}

// ---------------------------------------------------------------------------
// Kernel 1: router. One wave (32 lanes) per token. lane handles expert lane&7
// over a strided quarter of H; xor-shuffles reduce; softmax over 8; top-2.
// ---------------------------------------------------------------------------
__global__ void moe_router_kernel(const float* __restrict__ x,
                                  const float* __restrict__ gw,
                                  int* __restrict__ cursor,
                                  float* __restrict__ sumprob,
                                  int* __restrict__ list,
                                  float* __restrict__ wlist) {
  const int wave = threadIdx.x >> 5;
  const int lane = threadIdx.x & 31;
  const int t = blockIdx.x * 8 + wave;
  if (t >= T_) return;

  const int e = lane & 7;
  const float* xrow = x + (size_t)t * H_;
  const float* grow = gw + (size_t)e * H_;

  float acc = 0.0f;
  for (int h = (lane >> 3); h < H_; h += 4)
    acc += xrow[h] * grow[h];
  // combine the 4 lanes holding the same expert
  acc += __shfl_xor(acc, 8, 32);
  acc += __shfl_xor(acc, 16, 32);

  // softmax across the 8 experts (within each group of 8 lanes)
  float m = acc;
  m = fmaxf(m, __shfl_xor(m, 1, 32));
  m = fmaxf(m, __shfl_xor(m, 2, 32));
  m = fmaxf(m, __shfl_xor(m, 4, 32));
  float p = __expf(acc - m);
  float s = p;
  s += __shfl_xor(s, 1, 32);
  s += __shfl_xor(s, 2, 32);
  s += __shfl_xor(s, 4, 32);
  p = p / s;

  // every lane gathers all 8 probs of its group and scans top-2
  float pj[8];
  const int gbase = lane & 24;
#pragma unroll
  for (int j = 0; j < 8; ++j) pj[j] = __shfl(p, gbase + j, 32);

  int i1 = 0; float p1 = pj[0];
#pragma unroll
  for (int j = 1; j < 8; ++j) { if (pj[j] > p1) { p1 = pj[j]; i1 = j; } }
  int i2 = -1; float p2 = -1.0f;
#pragma unroll
  for (int j = 0; j < 8; ++j) { if (j != i1 && pj[j] > p2) { p2 = pj[j]; i2 = j; } }

  const float inv = 1.0f / (p1 + p2);
  if (lane == 0) {
    int pos1 = atomicAdd(&cursor[i1], 1);
    list[i1 * MAXP + pos1]  = t;
    wlist[i1 * MAXP + pos1] = p1 * inv;
    int pos2 = atomicAdd(&cursor[i2], 1);
    list[i2 * MAXP + pos2]  = t;
    wlist[i2 * MAXP + pos2] = p2 * inv;
  }
  if (lane < 8) atomicAdd(&sumprob[e], p);   // for load-balance loss
}

// ---------------------------------------------------------------------------
// Kernel 2: grouped sparse SwiGLU FFN, bf16 WMMA, fp32 accumulate.
// Grid: expert-major (L2 locality for the 50 MB/expert weight set).
// Workgroup: 256 threads (8 waves), 32-token tile, 128-wide F chunks.
// Activation tile is staged via GLOBAL_LOAD_ASYNC_TO_LDS_B128 (ASYNCcnt),
// then converted f32 -> bf16 once in LDS.
// ---------------------------------------------------------------------------
__global__ void __launch_bounds__(256)
moe_ffn_kernel(const float* __restrict__ x,
               const float* __restrict__ w1g,
               const float* __restrict__ w2g,
               const float* __restrict__ w3g,
               const int* __restrict__ cursor,
               const int* __restrict__ list,
               const float* __restrict__ wlist,
               float* __restrict__ y) {
  __shared__ __attribute__((aligned(16))) float  Xf[MT * H_];   // 128 KB async-staged fp32 tile
  __shared__ __attribute__((aligned(16))) __bf16 Xs[MT][HP];    // 66 KB bf16 activation tile
  __shared__ __attribute__((aligned(16))) __bf16 Hs[MT][FCP];   // silu(G)*U chunk (32x128)
  __shared__ float Wr[MT];
  __shared__ int   Tok[MT];

  const int eb   = blockIdx.x >> 7;          // expert (128 tiles per expert)
  const int tile = blockIdx.x & 127;
  const int cnt  = cursor[eb];
  const int row0 = tile * MT;
  if (row0 >= cnt) return;

  const int tid  = threadIdx.x;
  const int wave = tid >> 5;
  const int lane = tid & 31;
  const int laneN = lane & 15;
  const int half  = lane >> 4;

  if (tid < MT) {
    const int r = row0 + tid;
    const bool ok = r < cnt;
    Tok[tid] = ok ? list[eb * MAXP + r] : 0;
    Wr[tid]  = ok ? wlist[eb * MAXP + r] : 0.0f;
  }
  __syncthreads();

  // ---- Async-stage X tile: 32 rows x 1024 f32. Each thread issues one
  //      16-byte async LDS copy per row; per row the 256 threads cover 4 KB.
  {
#pragma unroll 4
    for (int i = 0; i < MT; ++i) {
      uint32_t ldsA = (uint32_t)(uintptr_t)(&Xf[i * H_ + tid * 4]);
      uint64_t gA   = (uint64_t)(uintptr_t)(x + (size_t)Tok[i] * H_ + tid * 4);
      asm volatile("global_load_async_to_lds_b128 %0, %1, off"
                   :: "v"(ldsA), "v"(gA) : "memory");
    }
    asm volatile("s_wait_asynccnt 0x0" ::: "memory");
  }
  __syncthreads();

  // ---- Convert staged f32 -> bf16 (each element exactly once) ----
  {
#pragma unroll 8
    for (int i = 0; i < MT; ++i) {
      float4 v = *(const float4*)(&Xf[i * H_ + tid * 4]);
      __bf16* dst = &Xs[i][tid * 4];
      dst[0] = (__bf16)v.x; dst[1] = (__bf16)v.y;
      dst[2] = (__bf16)v.z; dst[3] = (__bf16)v.w;
    }
  }
  __syncthreads();

  const float* W1 = w1g + (size_t)eb * F_ * H_;
  const float* W3 = w3g + (size_t)eb * F_ * H_;
  const float* W2 = w2g + (size_t)eb * H_ * F_;

  // Phase-1: each wave owns one 16-wide N-subtile of the 128-wide chunk and
  // computes BOTH 16-row M-subtiles of G and U (no weight duplication).
  // Persistent Y accumulators: this wave owns columns [wave*128, wave*128+128).
  v8f Yacc[2][8];
#pragma unroll
  for (int mi = 0; mi < 2; ++mi)
#pragma unroll
    for (int ni = 0; ni < 8; ++ni)
      Yacc[mi][ni] = (v8f){0.f, 0.f, 0.f, 0.f, 0.f, 0.f, 0.f, 0.f};

#pragma unroll 1
  for (int fc = 0; fc < F_; fc += FC) {
    // ---- Phase 1: G = X*W1^T, U = X*W3^T (two M-subtiles per wave) ----
    v8f G[2], U[2];
    G[0] = (v8f){0.f, 0.f, 0.f, 0.f, 0.f, 0.f, 0.f, 0.f};
    G[1] = G[0]; U[0] = G[0]; U[1] = G[0];
    const int nglob = fc + wave * 16 + laneN;              // row of w1/w3
    const float* w1row = W1 + (size_t)nglob * H_;
    const float* w3row = W3 + (size_t)nglob * H_;
#pragma unroll 2
    for (int k = 0; k < H_; k += 32) {
      v16bf B1 = cvt_b_frag(w1row + k + half * 16);
      v16bf B3 = cvt_b_frag(w3row + k + half * 16);
#pragma unroll
      for (int mi = 0; mi < 2; ++mi) {
        v16bf A = load_a_frag(&Xs[mi * 16][0], HP, k, lane);
        G[mi] = wmma_bf16(A, B1, G[mi]);
        U[mi] = wmma_bf16(A, B3, U[mi]);
      }
    }
    // h = silu(G) * U -> bf16 LDS (C/D layout: M = j + 8*half, N = laneN)
#pragma unroll
    for (int mi = 0; mi < 2; ++mi) {
#pragma unroll
      for (int j = 0; j < 8; ++j) {
        float g = G[mi][j], u = U[mi][j];
        float h = g * __builtin_amdgcn_rcpf(1.0f + __expf(-g)) * u;
        Hs[mi * 16 + j + 8 * half][wave * 16 + laneN] = (__bf16)h;
      }
    }
    __syncthreads();

    // ---- Phase 2: Y[:, wave slice] += Hs(32x128) @ W2slice^T ----
#pragma unroll
    for (int ks = 0; ks < FC; ks += 32) {
#pragma unroll
      for (int mi = 0; mi < 2; ++mi) {
        v16bf A2 = load_a_frag(&Hs[mi * 16][0], FCP, ks, lane);
#pragma unroll
        for (int ni = 0; ni < 8; ++ni) {
          const int nrow = wave * 128 + ni * 16 + laneN;   // row of w2 (H dim)
          v16bf B2 = cvt_b_frag(W2 + (size_t)nrow * F_ + fc + ks + half * 16);
          Yacc[mi][ni] = wmma_bf16(A2, B2, Yacc[mi][ni]);
        }
      }
    }
    __syncthreads();   // Hs reused next chunk
  }

  // ---- Epilogue: scale by routing weight, scatter-add into y ----
#pragma unroll
  for (int mi = 0; mi < 2; ++mi) {
#pragma unroll
    for (int j = 0; j < 8; ++j) {
      const int row = mi * 16 + j + 8 * half;
      const float wgt = Wr[row];
      if (wgt != 0.0f) {
        float* yrow = y + (size_t)Tok[row] * H_;
#pragma unroll
        for (int ni = 0; ni < 8; ++ni) {
          const int col = wave * 128 + ni * 16 + laneN;
          atomicAdd(&yrow[col], Yacc[mi][ni][j] * wgt);
        }
      }
    }
  }
}

// ---------------------------------------------------------------------------
// Kernel 3: load-balance loss = sum_e (sumprob[e]/T) * (count[e]/T)
// ---------------------------------------------------------------------------
__global__ void moe_finalize_kernel(const int* __restrict__ cursor,
                                    const float* __restrict__ sumprob,
                                    float* __restrict__ out) {
  if (threadIdx.x == 0 && blockIdx.x == 0) {
    float loss = 0.0f;
    for (int e = 0; e < E_; ++e)
      loss += sumprob[e] * (float)cursor[e];
    out[(size_t)T_ * H_] = loss / ((float)T_ * (float)T_);
  }
}

// ---------------------------------------------------------------------------
extern "C" void kernel_launch(void* const* d_in, const int* in_sizes, int n_in,
                              void* d_out, int out_size, void* d_ws, size_t ws_size,
                              hipStream_t stream) {
  const float* x  = (const float*)d_in[0];   // hidden_states [B,S,H]
  const float* gw = (const float*)d_in[1];   // gate_w [E,H]
  const float* w1 = (const float*)d_in[2];   // [E,F,H]
  const float* w2 = (const float*)d_in[3];   // [E,H,F]
  const float* w3 = (const float*)d_in[4];   // [E,F,H]
  float* out = (float*)d_out;                // y (T*H floats) + loss (1 float)

  // Workspace layout (floats/ints):
  //   [0..7]   cursor per expert (int)
  //   [8..15]  sumprob per expert (float)
  //   [16..)   list  int[E*MAXP]
  //   then     wlist float[E*MAXP]
  int*   cursor  = (int*)d_ws;
  float* sumprob = (float*)d_ws + 8;
  int*   list    = (int*)d_ws + 16;
  float* wlist   = (float*)d_ws + 16 + E_ * MAXP;

  const int nz = T_ * H_ + 1;
  moe_zero_kernel<<<(nz + 255) / 256, 256, 0, stream>>>(out, nz, (float*)d_ws);
  moe_router_kernel<<<T_ / 8, 256, 0, stream>>>(x, gw, cursor, sumprob, list, wlist);
  moe_ffn_kernel<<<E_ * (MAXP / MT), 256, 0, stream>>>(x, w1, w2, w3,
                                                       cursor, list, wlist, out);
  moe_finalize_kernel<<<1, 32, 0, stream>>>(cursor, sumprob, out);
}
